// TensorGNAN_43559558316167
// MI455X (gfx1250) — compile-verified, matrix operand order checked by reading, not run on hardware
//
#include <hip/hip_runtime.h>
#include <hip/hip_bf16.h>

typedef __attribute__((ext_vector_type(16))) _Float16 v16h;
typedef __attribute__((ext_vector_type(8)))  float    v8f;

#define N_NODES    2048
#define IN_CH      64
#define HID        16
#define OUT_C      8
#define NUM_GRAPHS 64
#define MSPLIT     16              // m-dimension split -> 128*16 = 2048 waves
#define MCHUNK     (N_NODES / MSPLIT)   // 128 m per wave

// ---------------------------------------------------------------------------
// Kernel 0: zero mf scratch and output (harness poisons them with 0xAA)
// ---------------------------------------------------------------------------
__global__ void zero_kernel(float* __restrict__ mf, float* __restrict__ out) {
    int i = blockIdx.x * blockDim.x + threadIdx.x;
    if (i < N_NODES * OUT_C) mf[i] = 0.0f;
    if (i < NUM_GRAPHS * OUT_C) out[i] = 0.0f;
}

// ---------------------------------------------------------------------------
// Kernel 1: g[n,c] = sum_f ( relu(x[n,f]*fs_w1[f,:]+fs_b1[f,:]) @ fs_w2[f] + fs_b2[f] )
// One thread per node n. Weights are uniform across threads -> scalar cached.
// ---------------------------------------------------------------------------
__global__ void g_kernel(const float* __restrict__ x,
                         const float* __restrict__ fs_w1,
                         const float* __restrict__ fs_b1,
                         const float* __restrict__ fs_w2,
                         const float* __restrict__ fs_b2,
                         float* __restrict__ g) {
    int n = blockIdx.x * blockDim.x + threadIdx.x;
    if (n >= N_NODES) return;
    float acc[OUT_C];
    #pragma unroll
    for (int c = 0; c < OUT_C; ++c) acc[c] = 0.0f;

    for (int f = 0; f < IN_CH; ++f) {
        float xv = x[n * IN_CH + f];
        const float* w1 = fs_w1 + f * HID;
        const float* b1 = fs_b1 + f * HID;
        const float* w2 = fs_w2 + f * HID * OUT_C;
        const float* b2 = fs_b2 + f * OUT_C;
        #pragma unroll
        for (int c = 0; c < OUT_C; ++c) acc[c] += b2[c];
        for (int h = 0; h < HID; ++h) {
            float hv = fmaxf(fmaf(xv, w1[h], b1[h]), 0.0f);
            #pragma unroll
            for (int c = 0; c < OUT_C; ++c)
                acc[c] = fmaf(hv, w2[h * OUT_C + c], acc[c]);
        }
    }
    #pragma unroll
    for (int c = 0; c < OUT_C; ++c) g[n * OUT_C + c] = acc[c];
}

// ---------------------------------------------------------------------------
// Kernel 2: pair stage as an f16 WMMA GEMM, fragments synthesized in-register.
//   mf[n,c] = sum_m sum_{k<32} A[n,(m,k)] * W[(m,k),c]
//   k<16 : A = valid*relu(d*rho_w1[k]+rho_b1[k]),  W = rho_w2[k,c]*g[m,c]
//   k==16: A = valid (via w1=0,b1=1 -> relu(1)=1), W = rho_b2[c]  *g[m,c]
//   k>16 : zero padding
//
// One wave / 16-row tile / m-chunk. Per iteration: one global_load_b128 of
// dist (4 m values), 4 back-to-back WMMAs ping-ponging two f32 accumulators
// so independent chains interleave with fragment-synthesis VALU. The wave's
// 4KB g-chunk is staged in LDS (coalesced b128 copy in, broadcast b32 out).
//
// f16 A-fragment lane layout (ISA 7.12.2, 16-bit A 16x32):
//   lane = row(0..15) + 16*grp ; halfword j -> k = j + 8*grp + (j>=8 ? 8 : 0)
// B (32x16) mirrors this with lane%16 = output column.
// ---------------------------------------------------------------------------
__global__ __launch_bounds__(32)
void pair_wmma_kernel(const float* __restrict__ dist,
                      const float* __restrict__ g,
                      const float* __restrict__ rho_w1,
                      const float* __restrict__ rho_b1,
                      const float* __restrict__ rho_w2,
                      const float* __restrict__ rho_b2,
                      float* __restrict__ mf) {
    __shared__ float gl[MCHUNK * OUT_C];      // 4 KB of the 320 KB/WGP LDS

    const int tile   = blockIdx.x;            // 0..127 : 16-row output tile
    const int mchunk = blockIdx.y;            // 0..MSPLIT-1
    const int lane   = threadIdx.x;           // 0..31 (one full wave32)
    const int row16  = lane & 15;
    const int grp    = lane >> 4;
    const int col    = row16;                 // B/D column index for this lane
    const int n_base = tile * 16;
    const int m_beg  = mchunk * MCHUNK;

    // Stage this wave's g chunk into LDS with coalesced 16B copies.
    {
        const float4* gsrc = (const float4*)(g + (size_t)m_beg * OUT_C);
        float4*       gdst = (float4*)gl;
        #pragma unroll
        for (int t = 0; t < (MCHUNK * OUT_C) / (4 * 32); ++t)   // 8 iters
            gdst[lane + t * 32] = gsrc[lane + t * 32];
    }
    __syncthreads();

    // Per-lane constant slot tables (k-mapping per ISA A-fragment layout).
    float w1s[16], b1s[16];
    v16h  cb;                                 // f16 B coefficients
    #pragma unroll
    for (int j = 0; j < 16; ++j) {
        int k = j + grp * 8 + ((j >= 8) ? 8 : 0);
        float w1v = 0.0f, b1v = 0.0f, cbv = 0.0f;
        if (k < HID)       { w1v = rho_w1[k]; b1v = rho_b1[k]; }
        else if (k == HID) { b1v = 1.0f; }    // relu(0*d+1)=1 -> valid indicator
        if (col < OUT_C) {
            if (k < HID)       cbv = rho_w2[k * OUT_C + col];
            else if (k == HID) cbv = rho_b2[col];
        }
        w1s[j] = w1v;
        b1s[j] = b1v;
        cb[j]  = (_Float16)cbv;
    }

    v8f acc0 = {};                            // two independent WMMA chains
    v8f acc1 = {};
    const float* drow = dist + (size_t)(n_base + row16) * N_NODES + m_beg;
    const float* gcol = gl + (col & (OUT_C - 1));

    for (int m = 0; m < MCHUNK; m += 4) {
        float4 d4 = *(const float4*)(drow + m);     // 16 MB dist read, once
        float dv[4] = {d4.x, d4.y, d4.z, d4.w};

        #pragma unroll
        for (int u = 0; u < 4; ++u) {
            float d     = dv[u];
            float vmask = (d >= 0.0f) ? 1.0f : 0.0f;

            v16h a;
            #pragma unroll
            for (int j = 0; j < 16; ++j) {
                float av = fmaxf(fmaf(d, w1s[j], b1s[j]), 0.0f) * vmask;
                a[j] = (_Float16)av;
            }

            _Float16 gh = (_Float16)gcol[(m + u) * OUT_C];  // LDS broadcast
            v16h gs;
            #pragma unroll
            for (int j = 0; j < 16; ++j) gs[j] = gh;
            v16h b = cb * gs;                               // v_pk_mul_f16

            if (u & 1)
                acc1 = __builtin_amdgcn_wmma_f32_16x16x32_f16(
                           false, a, false, b, (short)0, acc1, false, false);
            else
                acc0 = __builtin_amdgcn_wmma_f32_16x16x32_f16(
                           false, a, false, b, (short)0, acc0, false, false);
        }
    }

    // D layout: VGPR i holds row grp*8+i, column = lane&15. Only c<8 is real.
    if (col < OUT_C) {
        #pragma unroll
        for (int i = 0; i < 8; ++i) {
            int nr = n_base + grp * 8 + i;
            atomicAdd(&mf[nr * OUT_C + col], acc0[i] + acc1[i]);
        }
    }
}

// ---------------------------------------------------------------------------
// Kernel 3: graph pooling (segment sum) via global f32 atomics
// ---------------------------------------------------------------------------
__global__ void segsum_kernel(const float* __restrict__ mf,
                              const int* __restrict__ batch,
                              float* __restrict__ out) {
    int n = blockIdx.x * blockDim.x + threadIdx.x;
    if (n >= N_NODES) return;
    int b = batch[n];
    #pragma unroll
    for (int c = 0; c < OUT_C; ++c)
        atomicAdd(&out[b * OUT_C + c], mf[n * OUT_C + c]);
}

// ---------------------------------------------------------------------------
extern "C" void kernel_launch(void* const* d_in, const int* in_sizes, int n_in,
                              void* d_out, int out_size, void* d_ws, size_t ws_size,
                              hipStream_t stream) {
    const float* x       = (const float*)d_in[0];   // [2048,64]
    const float* dist    = (const float*)d_in[1];   // [2048,2048]
    const int*   batch   = (const int*)  d_in[2];   // [2048]
    const float* fs_w1   = (const float*)d_in[3];   // [64,16]
    const float* fs_b1   = (const float*)d_in[4];   // [64,16]
    const float* fs_w2   = (const float*)d_in[5];   // [64,16,8]
    const float* fs_b2   = (const float*)d_in[6];   // [64,8]
    const float* rho_w1  = (const float*)d_in[7];   // [16]
    const float* rho_b1  = (const float*)d_in[8];   // [16]
    const float* rho_w2  = (const float*)d_in[9];   // [16,8]
    const float* rho_b2  = (const float*)d_in[10];  // [8]
    float*       out     = (float*)d_out;           // [64,8]

    float* g  = (float*)d_ws;                       // [2048,8]  64 KB
    float* mf = g + N_NODES * OUT_C;                // [2048,8]  64 KB

    zero_kernel<<<(N_NODES * OUT_C + 255) / 256, 256, 0, stream>>>(mf, out);
    g_kernel<<<N_NODES / 256, 256, 0, stream>>>(x, fs_w1, fs_b1, fs_w2, fs_b2, g);
    pair_wmma_kernel<<<dim3(N_NODES / 16, MSPLIT), 32, 0, stream>>>(
        dist, g, rho_w1, rho_b1, rho_w2, rho_b2, mf);
    segsum_kernel<<<N_NODES / 256, 256, 0, stream>>>(mf, batch, out);
}
